// MLCV_VotingModule_47828755808280
// MI455X (gfx1250) — matrix operand order repaired and checked by reading, not compiled, not run on hardware
//
#include <hip/hip_runtime.h>
#include <hip/hip_bf16.h>

typedef __bf16 bf16;
typedef __attribute__((ext_vector_type(16))) __bf16 v16bf;
typedef __attribute__((ext_vector_type(8)))  __bf16 v8bf;
typedef __attribute__((ext_vector_type(8)))  float  v8f;

#define B_   64
#define C_   256
#define N_   1024
#define EPS_ 1e-5f

// ---------------- weight f32 -> bf16 pre-conversion (rows zero-padded to Mpad) ----------------
__global__ __launch_bounds__(256)
void cvt_w(const float* __restrict__ W, bf16* __restrict__ dst, int M)
{
  int idx = blockIdx.x * 256 + threadIdx.x;   // grid.x == Mpad, 256 cols per row
  int r = idx >> 8, c = idx & 255;
  dst[idx] = (r < M) ? (bf16)W[r * C_ + c] : (bf16)0.f;
}

// ---------------- WMMA bf16 GEMM:  Y[b] = W (M x 256, bf16, padded) * X[b] (256 x 1024) + bias --
#define BM 128
#define BN 128
#define BK 32
#define LDK 40   // padded LDS row (bf16 elems); 80B rows keep 16B chunk alignment

__global__ __launch_bounds__(256)
void gemm_wmma(const bf16* __restrict__ Wb, const float* __restrict__ X,
               const float* __restrict__ bias, float* __restrict__ Y,
               int M, long long strideY)
{
  __shared__ __align__(16) bf16 As[BM * LDK];
  __shared__ __align__(16) bf16 Bs[BN * LDK];

  const int tid   = threadIdx.x;
  const int lane  = tid & 31;        // wave32
  const int wave  = tid >> 5;        // 8 waves
  const int lrow  = lane & 15;       // M (A) / N (B,D) index within 16
  const int khalf = lane >> 4;       // which K-half of the fragment this lane holds

  const int m0 = blockIdx.x * BM;
  const int n0 = blockIdx.y * BN;
  const long long bz = blockIdx.z;
  const float* Xb = X + bz * (long long)(C_ * N_);
  float*       Yb = Y + bz * strideY;

  const int wm = wave & 3;   // 4 waves along M: 32 rows each
  const int wn = wave >> 2;  // 2 waves along N: 64 cols each

  v8f acc[2][4] = {};

  for (int k0 = 0; k0 < C_; k0 += BK) {
    // ---- A tile: async DMA bf16 weights -> LDS (128x32, 16B chunks, rows pre-padded) ----
    #pragma unroll
    for (int q = 0; q < 2; ++q) {
      int e  = q * 256 + tid;              // 512 chunks of 8 bf16
      int r  = e >> 2, c8 = e & 3;
      unsigned lds_off = (unsigned)(size_t)&As[r * LDK + c8 * 8];
      unsigned long long gp =
          (unsigned long long)(size_t)(Wb + ((m0 + r) << 8) + k0 + c8 * 8);
      asm volatile("global_load_async_to_lds_b128 %0, %1, off"
                   :: "v"(lds_off), "v"(gp) : "memory");
    }
    // ---- B tile: float4 loads along n, convert, transpose-scatter to Bs[n][k] ----
    #pragma unroll
    for (int q = 0; q < 4; ++q) {
      int e  = q * 256 + tid;              // 1024 float4 chunks
      int k  = e >> 5, n4 = e & 31;
      const float4 v = *(const float4*)&Xb[(k0 + k) * N_ + n0 + n4 * 4];
      Bs[(n4 * 4 + 0) * LDK + k] = (bf16)v.x;
      Bs[(n4 * 4 + 1) * LDK + k] = (bf16)v.y;
      Bs[(n4 * 4 + 2) * LDK + k] = (bf16)v.z;
      Bs[(n4 * 4 + 3) * LDK + k] = (bf16)v.w;
    }
    if (k0 + BK < C_)
      __builtin_prefetch(&Xb[(k0 + BK + (tid >> 7)) * N_ + n0 + (tid & 127)], 0, 1);

    asm volatile("s_wait_asynccnt 0x0" ::: "memory");
    __syncthreads();

    // ---- fragments per ISA 16-bit layouts ----
    v16bf a[2], b[4];
    #pragma unroll
    for (int ti = 0; ti < 2; ++ti) {       // A: elems 0..7 -> K=8h..8h+7 ; 8..15 -> K=8h+16..8h+23
      int row = wm * 32 + ti * 16 + lrow;
      union { v16bf v; v8bf h[2]; } u;
      u.h[0] = *(const v8bf*)&As[row * LDK + khalf * 8];
      u.h[1] = *(const v8bf*)&As[row * LDK + khalf * 8 + 16];
      a[ti] = u.v;
    }
    #pragma unroll
    for (int tj = 0; tj < 4; ++tj) {       // B: elems 0..7 -> K=8h..8h+7 ; 8..15 -> K=8h+16..8h+23
      int row = wn * 64 + tj * 16 + lrow;
      union { v16bf v; v8bf h[2]; } u;
      u.h[0] = *(const v8bf*)&Bs[row * LDK + khalf * 8];
      u.h[1] = *(const v8bf*)&Bs[row * LDK + khalf * 8 + 16];
      b[tj] = u.v;
    }

    #pragma unroll
    for (int ti = 0; ti < 2; ++ti)
      #pragma unroll
      for (int tj = 0; tj < 4; ++tj)
        acc[ti][tj] = __builtin_amdgcn_wmma_f32_16x16x32_bf16(
            false, a[ti], false, b[tj], (short)0, acc[ti][tj], false, false);

    __syncthreads();
  }

  // ---- store D: lane holds N = lane%16, M = 8*khalf + r ----
  #pragma unroll
  for (int ti = 0; ti < 2; ++ti) {
    int gmb = m0 + wm * 32 + ti * 16 + khalf * 8;
    #pragma unroll
    for (int tj = 0; tj < 4; ++tj) {
      int gn = n0 + wn * 64 + tj * 16 + lrow;
      #pragma unroll
      for (int r = 0; r < 8; ++r) {
        int gm = gmb + r;
        if (gm < M) {
          float bv = bias ? bias[gm] : 0.f;
          Yb[(long long)gm * N_ + gn] = acc[ti][tj][r] + bv;
        }
      }
    }
  }
}

// ---------------- fused: att reduction + Z GEMM + group-norm + residual ----------------
__global__ __launch_bounds__(256)
void cgnl_att_z_gn(const float* __restrict__ Tbuf, const float* __restrict__ Pbuf,
                   const float* __restrict__ Gbuf, const float* __restrict__ Wz,
                   const float* __restrict__ gamma, const float* __restrict__ beta,
                   const float* __restrict__ X, float* __restrict__ Y0)
{
  const int bg = blockIdx.x;
  const int b = bg >> 2, g = bg & 3;
  const int tid = threadIdx.x;

  __shared__ float wz[64 * 32];
  __shared__ float red[256], red2[256];
  __shared__ float s_att, s_mu, s_rs;

  for (int i = tid; i < 64 * 32; i += 256) wz[i] = Wz[g * (64 * 32) + i];

  const float* Pg = Pbuf + ((long long)b * 128 + g * 32) * N_;
  const float* Gg = Gbuf + ((long long)b * 128 + g * 32) * N_;
  float s = 0.f;
  for (int i = tid; i < 32 * N_; i += 256) s += Pg[i] * Gg[i];
  red[tid] = s; __syncthreads();
  for (int off = 128; off > 0; off >>= 1) {
    if (tid < off) red[tid] += red[tid + off];
    __syncthreads();
  }
  if (tid == 0) s_att = red[0];
  __syncthreads();
  const float att = s_att;

  const float* Tg = Tbuf + ((long long)b * 128 + g * 32) * N_;
  float* Zg = Y0 + ((long long)b * C_ + g * 64) * N_;
  float lsum = 0.f, lsq = 0.f;
  for (int i = tid; i < 64 * N_; i += 256) {
    int o = i >> 10, n = i & (N_ - 1);
    float acc = 0.f;
    #pragma unroll
    for (int c = 0; c < 32; ++c) acc = fmaf(wz[o * 32 + c], Tg[c * N_ + n], acc);
    float z = att * acc;
    Zg[i] = z;
    lsum += z; lsq += z * z;
  }
  red[tid] = lsum; red2[tid] = lsq; __syncthreads();
  for (int off = 128; off > 0; off >>= 1) {
    if (tid < off) { red[tid] += red[tid + off]; red2[tid] += red2[tid + off]; }
    __syncthreads();
  }
  if (tid == 0) {
    float mu = red[0] / 65536.f;
    float var = red2[0] / 65536.f - mu * mu;
    s_mu = mu; s_rs = rsqrtf(var + EPS_);
  }
  __syncthreads();
  const float mu = s_mu, rs = s_rs;

  const float* Xg = X + ((long long)b * C_ + g * 64) * N_;
  for (int i = tid; i < 64 * N_; i += 256) {
    int ch = g * 64 + (i >> 10);
    Zg[i] = (Zg[i] - mu) * rs * gamma[ch] + beta[ch] + Xg[i];
  }
}

// ---------------- BN stats: per-channel mean / rstd over (B, N) ----------------
__global__ __launch_bounds__(256)
void bn_stats(const float* __restrict__ H, float* __restrict__ mean, float* __restrict__ rstd)
{
  const int c = blockIdx.x;
  const int tid = threadIdx.x;
  __shared__ float red[256], red2[256];
  float s = 0.f, sq = 0.f;
  for (int i = tid; i < B_ * N_; i += 256) {
    int b = i >> 10, n = i & (N_ - 1);
    float v = H[((long long)b * C_ + c) * N_ + n];
    s += v; sq += v * v;
  }
  red[tid] = s; red2[tid] = sq; __syncthreads();
  for (int off = 128; off > 0; off >>= 1) {
    if (tid < off) { red[tid] += red[tid + off]; red2[tid] += red2[tid + off]; }
    __syncthreads();
  }
  if (tid == 0) {
    float mu = red[0] / (float)(B_ * N_);
    float var = red2[0] / (float)(B_ * N_) - mu * mu;
    mean[c] = mu;
    rstd[c] = rsqrtf(var + EPS_);
  }
}

// ---------------- BN normalize + affine + ReLU (in place) ----------------
__global__ __launch_bounds__(256)
void bn_relu(float* __restrict__ H, const float* __restrict__ mean, const float* __restrict__ rstd,
             const float* __restrict__ gamma, const float* __restrict__ beta)
{
  const long long total = (long long)B_ * C_ * N_;
  for (long long idx = (long long)blockIdx.x * blockDim.x + threadIdx.x; idx < total;
       idx += (long long)gridDim.x * blockDim.x) {
    int c = (int)((idx >> 10) & (C_ - 1));
    float v = H[idx];
    v = (v - mean[c]) * rstd[c] * gamma[c] + beta[c];
    H[idx] = fmaxf(v, 0.f);
  }
}

// ---------------- final scatter: vote_xyz / vote_feat ----------------
__global__ __launch_bounds__(256)
void vote_out(const float* __restrict__ H3, const float* __restrict__ seed_xyz,
              const float* __restrict__ seed_feat, float* __restrict__ out_xyz,
              float* __restrict__ out_feat)
{
  const long long total = (long long)B_ * 259 * N_;
  for (long long idx = (long long)blockIdx.x * blockDim.x + threadIdx.x; idx < total;
       idx += (long long)gridDim.x * blockDim.x) {
    long long b = idx / (259 * N_);
    int rem = (int)(idx - b * (259 * N_));
    int m = rem >> 10, n = rem & (N_ - 1);
    float v = H3[idx];
    if (m < 3) {
      long long o = (b * N_ + n) * 3 + m;
      out_xyz[o] = seed_xyz[o] + v;
    } else {
      int c = m - 3;
      long long o = (b * C_ + c) * N_ + n;
      out_feat[o] = seed_feat[o] + v;
    }
  }
}

extern "C" void kernel_launch(void* const* d_in, const int* in_sizes, int n_in,
                              void* d_out, int out_size, void* d_ws, size_t ws_size,
                              hipStream_t stream)
{
  const float* seed_xyz  = (const float*)d_in[0];
  const float* seed_feat = (const float*)d_in[1];
  const float* Wt  = (const float*)d_in[2];
  const float* Wp  = (const float*)d_in[3];
  const float* Wg  = (const float*)d_in[4];
  const float* Wz  = (const float*)d_in[5];
  const float* gng = (const float*)d_in[6];
  const float* gnb = (const float*)d_in[7];
  const float* W1  = (const float*)d_in[8];
  const float* b1  = (const float*)d_in[9];
  const float* bn1g = (const float*)d_in[10];
  const float* bn1b = (const float*)d_in[11];
  const float* W2  = (const float*)d_in[12];
  const float* b2  = (const float*)d_in[13];
  const float* bn2g = (const float*)d_in[14];
  const float* bn2b = (const float*)d_in[15];
  const float* W3  = (const float*)d_in[16];
  const float* b3  = (const float*)d_in[17];

  char* ws = (char*)d_ws;
  const size_t MB = 1024ull * 1024ull;
  // Layout (aliased; peak ~225MB):
  float* Tbuf = (float*)(ws + 0 * MB);     // (B,128,1024)  32MB   } dead after cgnl
  float* Pbuf = (float*)(ws + 32 * MB);    // 32MB                 }
  float* Gbuf = (float*)(ws + 64 * MB);    // 32MB                 }
  float* Y0   = (float*)(ws + 96 * MB);    // (B,256,1024) 64MB ; later reused as H2/Y2
  float* H1   = (float*)(ws + 160 * MB);   // (B,256,1024) 64MB  (H1 -> Y1 in place)
  float* H3   = (float*)(ws + 0 * MB);     // (B,259,1024) ~68MB over dead TPG region
  float* mean1 = (float*)(ws + 224 * MB);
  float* rstd1 = mean1 + 256;
  float* mean2 = rstd1 + 256;
  float* rstd2 = mean2 + 256;
  float* H2 = Y0;
  // bf16 weight staging (zero-padded rows)
  bf16* WtB = (bf16*)(ws + 224 * MB + 8192);
  bf16* WpB = WtB + 128 * 256;
  bf16* WgB = WpB + 128 * 256;
  bf16* W1B = WgB + 128 * 256;
  bf16* W2B = W1B + 256 * 256;
  bf16* W3B = W2B + 256 * 256;   // padded to 384 rows

  dim3 blk(256);

  // convert weights to bf16 once per launch
  cvt_w<<<dim3(128), blk, 0, stream>>>(Wt, WtB, 128);
  cvt_w<<<dim3(128), blk, 0, stream>>>(Wp, WpB, 128);
  cvt_w<<<dim3(128), blk, 0, stream>>>(Wg, WgB, 128);
  cvt_w<<<dim3(256), blk, 0, stream>>>(W1, W1B, 256);
  cvt_w<<<dim3(256), blk, 0, stream>>>(W2, W2B, 256);
  cvt_w<<<dim3(384), blk, 0, stream>>>(W3, W3B, 259);

  // T, P, G projections: 128x256 @ 256x1024 per batch (WMMA bf16)
  dim3 gproj(1, N_ / BN, B_);
  gemm_wmma<<<gproj, blk, 0, stream>>>(WtB, seed_feat, nullptr, Tbuf, 128, 128LL * N_);
  gemm_wmma<<<gproj, blk, 0, stream>>>(WpB, seed_feat, nullptr, Pbuf, 128, 128LL * N_);
  gemm_wmma<<<gproj, blk, 0, stream>>>(WgB, seed_feat, nullptr, Gbuf, 128, 128LL * N_);

  // att + Z + group-norm + residual -> Y0
  cgnl_att_z_gn<<<dim3(B_ * 4), blk, 0, stream>>>(Tbuf, Pbuf, Gbuf, Wz, gng, gnb, seed_feat, Y0);

  // layer 1
  dim3 g256(2, N_ / BN, B_);
  gemm_wmma<<<g256, blk, 0, stream>>>(W1B, Y0, b1, H1, 256, 256LL * N_);
  bn_stats<<<dim3(C_), blk, 0, stream>>>(H1, mean1, rstd1);
  bn_relu<<<dim3(8192), blk, 0, stream>>>(H1, mean1, rstd1, bn1g, bn1b);

  // layer 2
  gemm_wmma<<<g256, blk, 0, stream>>>(W2B, H1, b2, H2, 256, 256LL * N_);
  bn_stats<<<dim3(C_), blk, 0, stream>>>(H2, mean2, rstd2);
  bn_relu<<<dim3(8192), blk, 0, stream>>>(H2, mean2, rstd2, bn2g, bn2b);

  // layer 3 (M padded to 384 in W3B; stores guarded at 259)
  dim3 g259(3, N_ / BN, B_);
  gemm_wmma<<<g259, blk, 0, stream>>>(W3B, H2, b3, H3, 259, 259LL * N_);

  float* out_xyz  = (float*)d_out;
  float* out_feat = out_xyz + (long long)B_ * N_ * 3;
  vote_out<<<dim3(4096), blk, 0, stream>>>(H3, seed_xyz, seed_feat, out_xyz, out_feat);
}
